// FleetGNN_71794673320467
// MI455X (gfx1250) — matrix operand order repaired and checked by reading, not compiled, not run on hardware
//
#include <hip/hip_runtime.h>
#include <hip/hip_bf16.h>

typedef __attribute__((ext_vector_type(16))) _Float16 v16h;
typedef __attribute__((ext_vector_type(8)))  _Float16 v8h;
typedef __attribute__((ext_vector_type(8)))  float    v8f;
typedef __attribute__((ext_vector_type(4)))  float    v4f;
typedef __attribute__((ext_vector_type(4)))  unsigned int tdm_u32x4;
typedef __attribute__((ext_vector_type(8)))  int          tdm_i32x8;
typedef __attribute__((ext_vector_type(4)))  int          tdm_i32x4;

#define DEVN  100000
#define DEVE  1000000
#define DEVEE 1100000   // E + N (self loops appended)

__device__ inline float waveSum32(float v) {
  v += __shfl_xor(v, 1);
  v += __shfl_xor(v, 2);
  v += __shfl_xor(v, 4);
  v += __shfl_xor(v, 8);
  v += __shfl_xor(v, 16);
  return v;
}

__device__ inline void atomicAddF(float* p, float v) {
  __hip_atomic_fetch_add(p, v, __ATOMIC_RELAXED, __HIP_MEMORY_SCOPE_AGENT);
}

// float max via order-preserving integer atomics (init pattern: 0xFFFFFFFF)
__device__ inline void atomicMaxF(float* p, float v) {
  if (v >= 0.0f) atomicMax((int*)p, __float_as_int(v));
  else           atomicMin((unsigned int*)p, __float_as_uint(v));
}

// ---------------------------------------------------------------------------
// Tensor Data Mover: DMA a dense 2D f32 region (dim1 rows x dim0 cols, row
// stride stride0 elements) from global memory into LDS at byte offset ldsAddr.
// Descriptor per CDNA5 ISA ch.8: group0 = {flags, lds_addr, global_addr, type},
// group1 = {data_size, dims, tile dims, stride}, groups 2/3 = zero (2D tensor).
// This toolchain exposes the 6-arg builtin (extra zero-filled i32x8 group).
// ---------------------------------------------------------------------------
__device__ inline void tdm_load_2d_f32(const float* gsrc, unsigned ldsAddr,
                                       unsigned dim0, unsigned dim1,
                                       unsigned stride0) {
  unsigned long long ga = (unsigned long long)(const void*)gsrc;
  tdm_u32x4 g0;
  g0[0] = 1u;                                               // count=1 (valid), user mode
  g0[1] = ldsAddr;                                          // lds_addr (bytes)
  g0[2] = (unsigned)ga;                                     // global_addr[31:0]
  g0[3] = (unsigned)((ga >> 32) & 0x1FFFFFFull) | (2u << 30); // addr[56:32] | type=2
  tdm_i32x8 g1;
  g1[0] = (int)(2u << 16);                                  // data_size=4B, no multicast
  g1[1] = (int)((dim0 & 0xFFFFu) << 16);                    // tensor_dim0[15:0]
  g1[2] = (int)((dim0 >> 16) | ((dim1 & 0xFFFFu) << 16));   // dim0 hi | tensor_dim1 lo
  g1[3] = (int)((dim1 >> 16) | ((dim0 & 0xFFFFu) << 16));   // dim1 hi | tile_dim0
  g1[4] = (int)(dim1 & 0xFFFFu);                            // tile_dim1 (tile_dim2=0)
  g1[5] = (int)stride0;                                     // tensor_dim0_stride[31:0]
  g1[6] = 0;                                                // stride0 hi | dim1_stride lo
  g1[7] = 0;
  tdm_i32x4 z4 = {0, 0, 0, 0};
  tdm_i32x8 z8 = {0, 0, 0, 0, 0, 0, 0, 0};
  __builtin_amdgcn_tensor_load_to_lds(g0, g1, z4, z4, z8, 0);
}

// ---------------------------------------------------------------------------
// C(M x 128) = A(M x K) @ W(K x 128) [+ bias], K in {32,64}, M % 16 == 0.
// TDM stages the 16xK A-tile and the Kx128 W panel into LDS; 8 waves/block
// each compute one 16x16 tile with v_wmma_f32_16x16x32_f16; output staged in
// LDS and stored with coalesced b128 stores.
// ---------------------------------------------------------------------------
template <bool HALF_OUT>
__global__ void gemm_wmma(const float* __restrict__ A, const float* __restrict__ W,
                          const float* __restrict__ bias, void* __restrict__ Cout,
                          int K) {
  extern __shared__ float lds[];
  float* Als = lds;                     // 16 x K
  float* Wls = lds + 16 * K;            // K x 128
  float* Cst = Wls + (long)K * 128;     // 16 x 128 output staging

  const int lane = threadIdx.x & 31;
  const int wave = threadIdx.x >> 5;    // column tile 0..7
  const int r    = lane & 15;
  const int hh   = lane >> 4;           // lane half
  const long tileRow = blockIdx.x;
  const int col = wave * 16 + r;        // B column == C column

  if (wave == 0) {
    unsigned ldsBase = (unsigned)(unsigned long long)(void*)lds;
    tdm_load_2d_f32(A + tileRow * 16 * (long)K, ldsBase, (unsigned)K, 16u, (unsigned)K);
    tdm_load_2d_f32(W, ldsBase + (unsigned)(16 * K * 4), 128u, (unsigned)K, 128u);
    __builtin_amdgcn_s_wait_tensorcnt(0);
  }
  __syncthreads();

  v8f acc = {0.f, 0.f, 0.f, 0.f, 0.f, 0.f, 0.f, 0.f};
  for (int k0 = 0; k0 < K; k0 += 32) {
    v16h a, b;
    // A 16x32 f16 layout: VGPR j(0..3): K=2j+8h ; VGPR j(4..7): K=16+2(j-4)+8h
#pragma unroll
    for (int j = 0; j < 4; ++j) {
      int ka = k0 + 2 * j + 8 * hh;
      a[2 * j]     = (_Float16)Als[r * K + ka];
      a[2 * j + 1] = (_Float16)Als[r * K + ka + 1];
      int ka2 = k0 + 16 + 2 * j + 8 * hh;
      a[8 + 2 * j]     = (_Float16)Als[r * K + ka2];
      a[8 + 2 * j + 1] = (_Float16)Als[r * K + ka2 + 1];
    }
    // B 32x16 f16 layout: VGPR j(0..3): K=2j+8h ; VGPR j(4..7): K=16+2(j-4)+8h
#pragma unroll
    for (int j = 0; j < 4; ++j) {
      int kb = k0 + 2 * j + 8 * hh;
      b[2 * j]     = (_Float16)Wls[kb * 128 + col];
      b[2 * j + 1] = (_Float16)Wls[(kb + 1) * 128 + col];
      int kb2 = k0 + 16 + 2 * j + 8 * hh;
      b[8 + 2 * j]     = (_Float16)Wls[kb2 * 128 + col];
      b[8 + 2 * j + 1] = (_Float16)Wls[(kb2 + 1) * 128 + col];
    }
    acc = __builtin_amdgcn_wmma_f32_16x16x32_f16(false, a, false, b,
                                                 (short)0, acc, false, false);
  }
  const float bv = bias ? bias[col] : 0.0f;
#pragma unroll
  for (int i = 0; i < 8; ++i)                       // C/D layout: VGPR i -> M=i (+8 hi half)
    Cst[(i + 8 * hh) * 128 + col] = acc[i] + bv;
  __syncthreads();

  // cooperative coalesced store: thread t -> row t/16, 8-element segment t%16
  const int row = threadIdx.x >> 4;
  const int seg = threadIdx.x & 15;
  const float* src = Cst + row * 128 + seg * 8;
  const long gbase = (tileRow * 16 + row) * 128 + seg * 8;
  if (HALF_OUT) {
    v8h hv;
#pragma unroll
    for (int j = 0; j < 8; ++j) hv[j] = (_Float16)src[j];
    *(v8h*)((_Float16*)Cout + gbase) = hv;          // one b128 store
  } else {
    *(v4f*)((float*)Cout + gbase)     = *(const v4f*)src;
    *(v4f*)((float*)Cout + gbase + 4) = *(const v4f*)(src + 4);
  }
}

// ---------------------------------------------------------------------------
// Edge feature preprocessing
// ---------------------------------------------------------------------------
__global__ void reduce_ea(const float* __restrict__ ea, float* __restrict__ sums) {
  long stride = (long)gridDim.x * blockDim.x;
  float s0 = 0.f, s1 = 0.f;
  for (long e = (long)blockIdx.x * blockDim.x + threadIdx.x; e < DEVE; e += stride) {
    s0 += ea[2 * e];
    s1 += ea[2 * e + 1];
  }
  s0 = waveSum32(s0);
  s1 = waveSum32(s1);
  if ((threadIdx.x & 31) == 0) { atomicAddF(&sums[0], s0); atomicAddF(&sums[1], s1); }
}

__global__ void fill_efeat(const float* __restrict__ ea, const float* __restrict__ epw,
                           const float* __restrict__ epb, const float* __restrict__ sums,
                           float* __restrict__ efeat) {
  long e = (long)blockIdx.x * blockDim.x + threadIdx.x;
  if (e >= DEVEE) return;
  float a0, a1;
  if (e < DEVE) { a0 = ea[2 * e]; a1 = ea[2 * e + 1]; }
  else          { a0 = sums[0] * (1.f / DEVE); a1 = sums[1] * (1.f / DEVE); }  // mean fill
  float* o = efeat + e * 32;
#pragma unroll
  for (int j = 0; j < 32; ++j) o[j] = a0 * epw[j] + a1 * epw[32 + j] + epb[j];
}

// ---------------------------------------------------------------------------
// Pass A: attention logits + segment max.  One wave per edge; 8 lanes/head.
// ---------------------------------------------------------------------------
__global__ void edge_logits(const float* __restrict__ xl, const float* __restrict__ xr,
                            const _Float16* __restrict__ ef, const float* __restrict__ att,
                            const int* __restrict__ ei, float* __restrict__ alog,
                            float* __restrict__ amax) {
  const long e = (long)blockIdx.x * 8 + (threadIdx.x >> 5);
  if (e >= DEVEE) return;
  const int lane = threadIdx.x & 31;
  int s, d;
  if (e < DEVE) { s = ei[e]; d = ei[DEVE + e]; }
  else          { s = d = (int)(e - DEVE); }       // self loop
  const int h    = lane >> 3;
  const int base = h * 32 + (lane & 7) * 4;
  const float*    pl = xl + (long)s * 128 + base;
  const float*    pr = xr + (long)d * 128 + base;
  const _Float16* pe = ef + e * 128 + base;
  const float*    pa = att + base;
  float v = 0.f;
#pragma unroll
  for (int c = 0; c < 4; ++c) {
    float m = pl[c] + pr[c] + (float)pe[c];
    m = (m > 0.f) ? m : 0.2f * m;                  // leaky_relu(0.2)
    v += m * pa[c];
  }
  v += __shfl_xor(v, 1);
  v += __shfl_xor(v, 2);
  v += __shfl_xor(v, 4);                           // per-head partial sums
  if ((lane & 7) == 0) {
    alog[e * 4 + h] = v;
    atomicMaxF(&amax[(long)d * 4 + h], v);
  }
}

// ---------------------------------------------------------------------------
// Pass B: w = exp(a - amax[dst]); denom += w; outAcc[dst] += w * xl[src]
// ---------------------------------------------------------------------------
__global__ void edge_aggregate(const float* __restrict__ xl, const int* __restrict__ ei,
                               const float* __restrict__ alog, const float* __restrict__ amax,
                               float* __restrict__ denom, float* __restrict__ outAcc) {
  const long e = (long)blockIdx.x * 8 + (threadIdx.x >> 5);
  if (e >= DEVEE) return;
  const int lane = threadIdx.x & 31;
  int s, d;
  if (e < DEVE) { s = ei[e]; d = ei[DEVE + e]; }
  else          { s = d = (int)(e - DEVE); }
  const int h    = lane >> 3;
  const int base = h * 32 + (lane & 7) * 4;
  float w = __expf(alog[e * 4 + h] - amax[(long)d * 4 + h]);
  if ((lane & 7) == 0) atomicAddF(&denom[(long)d * 4 + h], w);
  const float* pl = xl + (long)s * 128 + base;
  float* po = outAcc + (long)d * 128 + base;
#pragma unroll
  for (int c = 0; c < 4; ++c) atomicAddF(&po[c], w * pl[c]);
}

// ---------------------------------------------------------------------------
// Pass C: head mean + conv bias + LayerNorm + ReLU. One wave per node.
// ---------------------------------------------------------------------------
__global__ void node_finalize(const float* __restrict__ outAcc, const float* __restrict__ denom,
                              const float* __restrict__ cb, const float* __restrict__ lng,
                              const float* __restrict__ lnb, float* __restrict__ hout) {
  const long n = (long)blockIdx.x * 8 + (threadIdx.x >> 5);
  if (n >= DEVN) return;
  const int c = threadIdx.x & 31;
  float o = 0.f;
#pragma unroll
  for (int h = 0; h < 4; ++h) o += outAcc[n * 128 + h * 32 + c] / denom[n * 4 + h];
  o = 0.25f * o + cb[c];
  float mu  = waveSum32(o) * (1.f / 32.f);
  float dv  = o - mu;
  float var = waveSum32(dv * dv) * (1.f / 32.f);
  float y = dv * rsqrtf(var + 1e-5f) * lng[c] + lnb[c];
  hout[n * 32 + c] = fmaxf(y, 0.f);
}

// ---------------------------------------------------------------------------
// Heads
// ---------------------------------------------------------------------------
__global__ void pool_mean(const float* __restrict__ h2, float* __restrict__ gsum) {
  int c = threadIdx.x & 31;
  long t = (long)blockIdx.x * blockDim.x + threadIdx.x;
  long rstride = ((long)gridDim.x * blockDim.x) >> 5;
  float s = 0.f;
  for (long r = t >> 5; r < DEVN; r += rstride) s += h2[r * 32 + c];
  atomicAddF(&gsum[c], s);
}

__global__ void graph_heads(const float* __restrict__ gsum,
                            const float* fw1, const float* fb1, const float* fw2, const float* fb2,
                            const float* iw1, const float* ib1, const float* iw2, const float* ib2,
                            float* __restrict__ out) {
  __shared__ float gs[32];
  int j = threadIdx.x;
  gs[j] = gsum[j] * (1.f / DEVN);
  __syncthreads();
  float t = fb1[j];
  for (int k = 0; k < 32; ++k) t += gs[k] * fw1[k * 32 + j];
  t = fmaxf(t, 0.f);
  float pf = waveSum32(t * fw2[j]);
  float pi = 0.f;
  if (j < 16) {
    float u = ib1[j];
    for (int k = 0; k < 32; ++k) u += gs[k] * iw1[k * 16 + j];
    pi = fmaxf(u, 0.f) * iw2[j];
  }
  pi = waveSum32(pi);
  if (j == 0) {
    out[0] = 1.f / (1.f + __expf(-(pf + fb2[0])));   // fleet_health
    out[3] = 1.f / (1.f + __expf(-(pi + ib2[0])));   // imbalance
  }
}

__global__ void node_scores(const float* __restrict__ h2,
                            const float* aw1, const float* ab1, const float* aw2, const float* ab2,
                            unsigned long long* __restrict__ key) {
  long n = (long)blockIdx.x * blockDim.x + threadIdx.x;
  if (n >= DEVN) return;
  const float* hr = h2 + n * 32;
  float acc = ab2[0];
  for (int j = 0; j < 16; ++j) {
    float u = ab1[j];
    for (int k = 0; k < 32; ++k) u += hr[k] * aw1[k * 16 + j];
    acc += fmaxf(u, 0.f) * aw2[j];
  }
  float score = 1.f / (1.f + __expf(-acc));          // in (0,1): bits order-preserving
  unsigned long long k = ((unsigned long long)__float_as_uint(score) << 32)
                       | (unsigned long long)(0xFFFFFFFFu - (unsigned)n);
  atomicMax(key, k);                                  // ties -> smallest index wins
}

__global__ void extract_argmax(const unsigned long long* __restrict__ key,
                               float* __restrict__ out) {
  unsigned long long k = *key;
  out[1] = (float)(0xFFFFFFFFu - (unsigned)(k & 0xFFFFFFFFull));  // outlier_idx
  out[2] = __uint_as_float((unsigned)(k >> 32));                  // outlier_score
}

// ---------------------------------------------------------------------------
extern "C" void kernel_launch(void* const* d_in, const int* in_sizes, int n_in,
                              void* d_out, int out_size, void* d_ws, size_t ws_size,
                              hipStream_t stream) {
  (void)in_sizes; (void)n_in; (void)out_size; (void)ws_size;
  const float* x    = (const float*)d_in[0];
  const int*   ei   = (const int*)d_in[1];
  const float* ea   = (const float*)d_in[2];
  const float* epw  = (const float*)d_in[3];
  const float* epb  = (const float*)d_in[4];
  const float* Wl1  = (const float*)d_in[5];
  const float* bl1  = (const float*)d_in[6];
  const float* Wr1  = (const float*)d_in[7];
  const float* br1  = (const float*)d_in[8];
  const float* We1  = (const float*)d_in[9];
  const float* att1 = (const float*)d_in[10];
  const float* cb1  = (const float*)d_in[11];
  const float* Wl2  = (const float*)d_in[12];
  const float* bl2  = (const float*)d_in[13];
  const float* Wr2  = (const float*)d_in[14];
  const float* br2  = (const float*)d_in[15];
  const float* We2  = (const float*)d_in[16];
  const float* att2 = (const float*)d_in[17];
  const float* cb2  = (const float*)d_in[18];
  const float* ln1g = (const float*)d_in[19];
  const float* ln1b = (const float*)d_in[20];
  const float* ln2g = (const float*)d_in[21];
  const float* ln2b = (const float*)d_in[22];
  const float* fw1  = (const float*)d_in[23];
  const float* fb1  = (const float*)d_in[24];
  const float* fw2  = (const float*)d_in[25];
  const float* fb2  = (const float*)d_in[26];
  const float* aw1  = (const float*)d_in[27];
  const float* ab1  = (const float*)d_in[28];
  const float* aw2  = (const float*)d_in[29];
  const float* ab2  = (const float*)d_in[30];
  const float* iw1  = (const float*)d_in[31];
  const float* ib1  = (const float*)d_in[32];
  const float* iw2  = (const float*)d_in[33];
  const float* ib2  = (const float*)d_in[34];

  char* ws = (char*)d_ws;
  size_t off = 0;
  auto alloc = [&](size_t bytes) { size_t o = off; off += (bytes + 255) & ~(size_t)255; return o; };
  float*    efeat = (float*)(ws + alloc((size_t)DEVEE * 32 * 4));
  _Float16* ef    = (_Float16*)(ws + alloc((size_t)DEVEE * 128 * 2));
  float*    xl    = (float*)(ws + alloc((size_t)DEVN * 128 * 4));
  float*    xr    = (float*)(ws + alloc((size_t)DEVN * 128 * 4));
  float*    alog  = (float*)(ws + alloc((size_t)DEVEE * 4 * 4));
  float*    amax  = (float*)(ws + alloc((size_t)DEVN * 4 * 4));
  float*    denom = (float*)(ws + alloc((size_t)DEVN * 4 * 4));
  float*    oacc  = (float*)(ws + alloc((size_t)DEVN * 128 * 4));
  float*    h1    = (float*)(ws + alloc((size_t)DEVN * 32 * 4));
  float*    h2    = (float*)(ws + alloc((size_t)DEVN * 32 * 4));
  float*    sums  = (float*)(ws + alloc(2 * 4));
  float*    gsum  = (float*)(ws + alloc(32 * 4));
  unsigned long long* key = (unsigned long long*)(ws + alloc(8));

  // dynamic LDS: A tile (16xK) + W panel (Kx128) + C staging (16x128), f32
  const size_t smem32 = (size_t)(16 * 32 + 32 * 128 + 16 * 128) * 4;  // K=32
  const size_t smem64 = (size_t)(16 * 64 + 64 * 128 + 16 * 128) * 4;  // K=64

  // edge projection + self-loop mean fill
  (void)hipMemsetAsync(sums, 0, 2 * 4, stream);
  reduce_ea<<<256, 256, 0, stream>>>(ea, sums);
  fill_efeat<<<(DEVEE + 255) / 256, 256, 0, stream>>>(ea, epw, epb, sums, efeat);

  // ---- layer 1 ----
  gemm_wmma<true ><<<dim3(DEVEE / 16), 256, smem32, stream>>>(efeat, We1, nullptr, ef, 32);
  gemm_wmma<false><<<dim3(DEVN / 16), 256, smem64, stream>>>(x, Wl1, bl1, xl, 64);
  gemm_wmma<false><<<dim3(DEVN / 16), 256, smem64, stream>>>(x, Wr1, br1, xr, 64);
  (void)hipMemsetAsync(amax, 0xFF, (size_t)DEVN * 4 * 4, stream);
  (void)hipMemsetAsync(denom, 0, (size_t)DEVN * 4 * 4, stream);
  (void)hipMemsetAsync(oacc, 0, (size_t)DEVN * 128 * 4, stream);
  edge_logits<<<DEVEE / 8, 256, 0, stream>>>(xl, xr, ef, att1, ei, alog, amax);
  edge_aggregate<<<DEVEE / 8, 256, 0, stream>>>(xl, ei, alog, amax, denom, oacc);
  node_finalize<<<DEVN / 8, 256, 0, stream>>>(oacc, denom, cb1, ln1g, ln1b, h1);

  // ---- layer 2 ----
  gemm_wmma<true ><<<dim3(DEVEE / 16), 256, smem32, stream>>>(efeat, We2, nullptr, ef, 32);
  gemm_wmma<false><<<dim3(DEVN / 16), 256, smem32, stream>>>(h1, Wl2, bl2, xl, 32);
  gemm_wmma<false><<<dim3(DEVN / 16), 256, smem32, stream>>>(h1, Wr2, br2, xr, 32);
  (void)hipMemsetAsync(amax, 0xFF, (size_t)DEVN * 4 * 4, stream);
  (void)hipMemsetAsync(denom, 0, (size_t)DEVN * 4 * 4, stream);
  (void)hipMemsetAsync(oacc, 0, (size_t)DEVN * 128 * 4, stream);
  edge_logits<<<DEVEE / 8, 256, 0, stream>>>(xl, xr, ef, att2, ei, alog, amax);
  edge_aggregate<<<DEVEE / 8, 256, 0, stream>>>(xl, ei, alog, amax, denom, oacc);
  node_finalize<<<DEVN / 8, 256, 0, stream>>>(oacc, denom, cb2, ln2g, ln2b, h2);

  // ---- heads ----
  (void)hipMemsetAsync(gsum, 0, 32 * 4, stream);
  (void)hipMemsetAsync(key, 0, 8, stream);
  pool_mean<<<256, 256, 0, stream>>>(h2, gsum);
  graph_heads<<<1, 32, 0, stream>>>(gsum, fw1, fb1, fw2, fb2, iw1, ib1, iw2, ib2, (float*)d_out);
  node_scores<<<(DEVN + 255) / 256, 256, 0, stream>>>(h2, aw1, ab1, aw2, ab2, key);
  extract_argmax<<<1, 1, 0, stream>>>(key, (float*)d_out);
}
